// MyLSTM_65515431133364
// MI455X (gfx1250) — matrix operand order repaired and checked by reading, not compiled, not run on hardware
//
#include <hip/hip_runtime.h>

typedef __attribute__((ext_vector_type(16))) __bf16 v16bf;
typedef __attribute__((ext_vector_type(8)))  float  v8f;

#define HID   64
#define NG    256          // 4*HID gate width
#define DIM   512
#define TSEQ  9
#define BATCH 4096
#define MROWS (BATCH * TSEQ)   // 36864

// fp32 -> bf16 via hardware convert (RNE), returned as raw bits
__device__ __forceinline__ unsigned short f2bf(float f) {
  union { __bf16 b; unsigned short u; } v;
  v.b = (__bf16)f;
  return v.u;
}

// assemble a 16-element bf16 fragment from two contiguous 8-element runs
__device__ __forceinline__ v16bf load_frag(const unsigned short* p0,
                                           const unsigned short* p1) {
  union { uint4 q[2]; v16bf v; } u;
  u.q[0] = *(const uint4*)p0;
  u.q[1] = *(const uint4*)p1;
  return u.v;
}

// build a bf16 A-fragment from two 8-float runs held in LDS (v_cvt_pk_bf16_f32)
__device__ __forceinline__ v16bf cvt_frag(const float* p0, const float* p1) {
  union { __bf16 e[16]; v16bf v; } u;
  const float4 a0 = ((const float4*)p0)[0];
  const float4 a1 = ((const float4*)p0)[1];
  const float4 b0 = ((const float4*)p1)[0];
  const float4 b1 = ((const float4*)p1)[1];
  u.e[0]  = (__bf16)a0.x; u.e[1]  = (__bf16)a0.y;
  u.e[2]  = (__bf16)a0.z; u.e[3]  = (__bf16)a0.w;
  u.e[4]  = (__bf16)a1.x; u.e[5]  = (__bf16)a1.y;
  u.e[6]  = (__bf16)a1.z; u.e[7]  = (__bf16)a1.w;
  u.e[8]  = (__bf16)b0.x; u.e[9]  = (__bf16)b0.y;
  u.e[10] = (__bf16)b0.z; u.e[11] = (__bf16)b0.w;
  u.e[12] = (__bf16)b1.x; u.e[13] = (__bf16)b1.y;
  u.e[14] = (__bf16)b1.z; u.e[15] = (__bf16)b1.w;
  return u.v;
}

// generic LDS pointer -> raw 32-bit LDS byte address (flat aperture: addr[31:0])
__device__ __forceinline__ unsigned lds32(const void* p) {
  return (unsigned)(unsigned long long)p;
}

// 16-byte async copy global -> LDS (tracked on ASYNCcnt, bypasses VGPRs)
__device__ __forceinline__ void async_cp16(unsigned ldsAddr, const void* gAddr) {
  asm volatile("global_load_async_to_lds_b128 %0, %1, off"
               :: "v"(ldsAddr), "v"(gAddr) : "memory");
}

__device__ __forceinline__ float sigmoidf(float x) {
  return 1.0f / (1.0f + __expf(-x));
}

// ---------------------------------------------------------------------------
// Kernel 1: convert + transpose weights to bf16 [N][K]
// ---------------------------------------------------------------------------
__global__ __launch_bounds__(256) void lstm_prep_w(
    const float* __restrict__ K,
    unsigned short* __restrict__ WxT,
    unsigned short* __restrict__ WhT) {
  int i = blockIdx.x * 256 + threadIdx.x;      // 576 * 256 = 147456
  if (i < NG * DIM) {
    int n = i / DIM, k = i % DIM;
    WxT[i] = f2bf(K[(size_t)k * NG + n]);
  } else {
    int j = i - NG * DIM;
    int n = j / HID, k = j % HID;
    WhT[j] = f2bf(K[(size_t)(DIM + k) * NG + n]);
  }
}

// ---------------------------------------------------------------------------
// Kernel 2: XG = x @ Wx  (bf16 WMMA). Block tile 128x128, wave tile 32x64.
// Async global->LDS staging, triple-buffered, 2 K-tiles in flight, ONE
// barrier per K-iteration. A staged as fp32, converted at fragment build.
// ---------------------------------------------------------------------------
__global__ __launch_bounds__(256) void lstm_gemm_xw(
    const float* __restrict__ X,
    const unsigned short* __restrict__ WxT,
    float* __restrict__ XG) {
  __shared__ float          As[3][128 * 36];   // fp32 A tiles, stride 36
  __shared__ unsigned short Bs[3][128 * 56];   // bf16 B tiles, stride 56

  const int tid   = threadIdx.x;
  const int lane  = tid & 31;
  const int wave  = tid >> 5;
  const int wM    = wave >> 1;              // 0..3 -> M offset wM*32
  const int wN    = wave & 1;               // 0..1 -> N offset wN*64
  const int row16 = lane & 15;
  const int hi    = lane >> 4;
  const int kbA   = hi << 3;                // A frag: runs at kbA, kbA+16
  const int kbB   = hi << 4;                // unused; B frag also uses runs at kbA, kbA+16
  const int bN    = blockIdx.x;             // 0..1
  const int bM    = blockIdx.y;             // 0..287

  const float*          xBase = X   + (size_t)bM * 128 * DIM;
  const unsigned short* wBase = WxT + (size_t)bN * 128 * DIM;

  // staging roles: 2 threads per row, 16 elements each
  const int sr = tid >> 1, sh = tid & 1;
  const float*          aG = xBase + (size_t)sr * DIM + sh * 16;
  const unsigned short* bG = wBase + (size_t)sr * DIM + sh * 16;

  // 6 async b128 per thread per K-tile (4 for A fp32, 2 for B bf16)
  auto issue_tile = [&](int tix) {
    const int b = tix % 3;
    const float*          ag = aG + tix * 32;
    const unsigned short* bg = bG + tix * 32;
    const unsigned aL = lds32(&As[b][sr * 36 + sh * 16]);
    const unsigned bL = lds32(&Bs[b][sr * 56 + sh * 16]);
#pragma unroll
    for (int j = 0; j < 4; ++j) async_cp16(aL + j * 16, ag + j * 4);
#pragma unroll
    for (int j = 0; j < 2; ++j) async_cp16(bL + j * 16, bg + j * 8);
  };

  issue_tile(0);
  issue_tile(1);

  v8f acc[2][4] = {};

  for (int it = 0; it < DIM / 32; ++it) {
    // tile `it` landed when <=6 async ops (newest tile) remain outstanding
    if (it < 15) asm volatile("s_wait_asynccnt 0x6" ::: "memory");
    else         asm volatile("s_wait_asynccnt 0x0" ::: "memory");
    __syncthreads();   // all waves' async writes for tile `it` are in LDS

    const int bf = it % 3;
    const float*          Ab = As[bf];
    const unsigned short* Bb = Bs[bf];

    v16bf a[2], b[4];
#pragma unroll
    for (int mi = 0; mi < 2; ++mi) {
      const float* ap = Ab + (wM * 32 + mi * 16 + row16) * 36 + kbA;
      a[mi] = cvt_frag(ap, ap + 16);
    }
#pragma unroll
    for (int ni = 0; ni < 4; ++ni) {
      const unsigned short* bp = Bb + (wN * 64 + ni * 16 + row16) * 56;
      b[ni] = load_frag(bp + kbA, bp + kbA + 16);
    }
#pragma unroll
    for (int mi = 0; mi < 2; ++mi)
#pragma unroll
      for (int ni = 0; ni < 4; ++ni)
        acc[mi][ni] = __builtin_amdgcn_wmma_f32_16x16x32_bf16(
            false, a[mi], false, b[ni], (short)0, acc[mi][ni], false, false);

    // prefetch tile it+2 into the buffer whose readers all passed the barrier
    if (it + 2 < DIM / 32) issue_tile(it + 2);
  }

  // ---- write D fragments (C/D layout: lanes 0-15 rows 0-7, 16-31 rows 8-15)
#pragma unroll
  for (int mi = 0; mi < 2; ++mi) {
    const int rbase = bM * 128 + wM * 32 + mi * 16 + hi * 8;
#pragma unroll
    for (int ni = 0; ni < 4; ++ni) {
      const int col = bN * 128 + wN * 64 + ni * 16 + row16;
#pragma unroll
      for (int r = 0; r < 8; ++r)
        XG[(size_t)(rbase + r) * NG + col] = acc[mi][ni][r];
    }
  }
}

// ---------------------------------------------------------------------------
// Kernel 3: recurrent scan. 32 batch rows / block, 128 blocks.
// Wh fragments resident in registers across all 9 steps; h@Wh via WMMA into
// LDS gates buffer; fused elementwise adds xg + bias, sigmoid/tanh, c in regs.
// ---------------------------------------------------------------------------
__global__ __launch_bounds__(256) void lstm_scan(
    const float* __restrict__ XG,
    const unsigned short* __restrict__ WhT,
    const float* __restrict__ bias,
    float* __restrict__ OUT) {
  __shared__ float          gBuf[32 * 260];   // gates, stride-padded
  __shared__ unsigned short hBuf[32 * 72];    // h in bf16, stride-padded

  const int tid   = threadIdx.x;
  const int lane  = tid & 31;
  const int wave  = tid >> 5;                 // wave owns gate cols [32w,32w+32)
  const int row16 = lane & 15;
  const int hi    = lane >> 4;
  const int kbA   = hi << 3;
  const int kbB   = hi << 4;
  const int rBase = blockIdx.x * 32;

  // preload recurrent weight fragments: 2 N-tiles x 2 K-chunks per wave
  v16bf bfrag[2][2];
#pragma unroll
  for (int ni = 0; ni < 2; ++ni) {
    const unsigned short* p = WhT + (size_t)(wave * 32 + ni * 16 + row16) * HID;
#pragma unroll
    for (int kc = 0; kc < 2; ++kc)
      bfrag[ni][kc] = load_frag(p + kc * 32 + kbA, p + kc * 32 + kbA + 16);
  }

  for (int i = tid; i < 32 * 72; i += 256) hBuf[i] = 0;   // h0 = 0

  float c[8];
#pragma unroll
  for (int i = 0; i < 8; ++i) c[i] = 0.f;                 // c0 = 0

  const int   ehid  = tid & 63;
  const int   erow0 = tid >> 6;
  const float bI = bias[ehid];
  const float bJ = bias[ehid + 64];
  const float bF = bias[ehid + 128] + 1.0f;               // FORGET_BIAS folded
  const float bO = bias[ehid + 192];

  __syncthreads();

  for (int t = 0; t < TSEQ; ++t) {
    // ---- gates = h @ Wh  (8 WMMAs / wave / step) ----
#pragma unroll
    for (int mi = 0; mi < 2; ++mi) {
      const unsigned short* hp = hBuf + (mi * 16 + row16) * 72;
      v16bf a0 = load_frag(hp + kbA,      hp + kbA + 16);
      v16bf a1 = load_frag(hp + 32 + kbA, hp + 32 + kbA + 16);
#pragma unroll
      for (int ni = 0; ni < 2; ++ni) {
        v8f d = {};
        d = __builtin_amdgcn_wmma_f32_16x16x32_bf16(
            false, a0, false, bfrag[ni][0], (short)0, d, false, false);
        d = __builtin_amdgcn_wmma_f32_16x16x32_bf16(
            false, a1, false, bfrag[ni][1], (short)0, d, false, false);
        const int col = wave * 32 + ni * 16 + row16;
        const int rw  = mi * 16 + hi * 8;
#pragma unroll
        for (int r = 0; r < 8; ++r)
          gBuf[(rw + r) * 260 + col] = d[r];
      }
    }
    __syncthreads();

    // ---- fused elementwise: + xg + bias, nonlinearities, state update ----
#pragma unroll
    for (int p = 0; p < 8; ++p) {
      const int row  = erow0 + p * 4;
      const int grow = rBase + row;
      const float* xr = XG + ((size_t)grow * TSEQ + t) * NG;
      const float gi = gBuf[row * 260 + ehid]       + xr[ehid]       + bI;
      const float gj = gBuf[row * 260 + ehid + 64]  + xr[ehid + 64]  + bJ;
      const float gf = gBuf[row * 260 + ehid + 128] + xr[ehid + 128] + bF;
      const float go = gBuf[row * 260 + ehid + 192] + xr[ehid + 192] + bO;
      const float cn = sigmoidf(gf) * c[p] + sigmoidf(gi) * tanhf(gj);
      const float hn = sigmoidf(go) * tanhf(cn);
      c[p] = cn;
      OUT[((size_t)grow * TSEQ + t) * HID + ehid] = hn;
      hBuf[row * 72 + ehid] = f2bf(hn);
    }
    __syncthreads();
  }
}

// ---------------------------------------------------------------------------
extern "C" void kernel_launch(void* const* d_in, const int* in_sizes, int n_in,
                              void* d_out, int out_size, void* d_ws, size_t ws_size,
                              hipStream_t stream) {
  const float* x    = (const float*)d_in[0];   // (4096, 9, 512) fp32
  const float* kern = (const float*)d_in[1];   // (576, 256)     fp32
  const float* bias = (const float*)d_in[2];   // (256,)         fp32
  float* out        = (float*)d_out;           // (4096, 9, 64)  fp32

  // workspace layout
  float* XG = (float*)d_ws;                                         // 36864*256 f32
  unsigned short* WxT = (unsigned short*)(XG + (size_t)MROWS * NG); // 256*512 bf16
  unsigned short* WhT = WxT + NG * DIM;                             // 256*64  bf16

  lstm_prep_w<<<(NG * DIM + NG * HID) / 256, 256, 0, stream>>>(kern, WxT, WhT);
  lstm_gemm_xw<<<dim3(NG / 128, MROWS / 128), 256, 0, stream>>>(x, WxT, XG);
  lstm_scan<<<BATCH / 32, 256, 0, stream>>>(XG, WhT, bias, out);
}
